// FNO3d_44856638439812
// MI455X (gfx1250) — compile-verified, matrix operand order checked by reading, not run on hardware
//
#include <hip/hip_runtime.h>
#include <hip/hip_bf16.h>
#include <math.h>

// ---------------------------------------------------------------------------
// FNO3d on gfx1250 (MI455X): corner-mode partial-DFT factorization.
// All transform stages + pointwise conv use v_wmma_f32_16x16x32_bf16
// (bf16 inputs, f32 accumulate). wave32 throughout.
// ---------------------------------------------------------------------------

typedef __attribute__((ext_vector_type(16))) __bf16 v16bf;
typedef __attribute__((ext_vector_type(8)))  float  v8f;

#define NVOX (64 * 64 * 64) /* 262144 */

__device__ __forceinline__ v8f vzero() {
  v8f z;
#pragma unroll
  for (int i = 0; i < 8; ++i) z[i] = 0.f;
  return z;
}

__device__ __forceinline__ v8f wmma_bf16(v16bf a, v16bf b, v8f c) {
  // D = A(16x32 bf16) x B(32x16 bf16) + C(16x16 f32)
  return __builtin_amdgcn_wmma_f32_16x16x32_bf16(false, a, false, b, (short)0, c,
                                                 false, false);
}

// ---- fragment builders -----------------------------------------------------
// A-matrix (16x32, 16-bit) per ISA 7.12.2:
//   lanes 0-15 : row M=lane,  vgprs hold K = {0..7, 16..23}
//   lanes 16-31: row M=lane-16, K = {8..15, 24..31}
__device__ __forceinline__ v16bf frag_a_f32(const float* __restrict__ src, int ld) {
  int lane = threadIdx.x & 31, half = lane >> 4, m = lane & 15;
  const float* r = src + m * ld + half * 8;
  v16bf a;
#pragma unroll
  for (int j = 0; j < 8; ++j) { a[j] = (__bf16)r[j]; a[8 + j] = (__bf16)r[16 + j]; }
  return a;
}
// A with only K=0..15 valid (K padded with zeros to 32)
__device__ __forceinline__ v16bf frag_a_f32_k16(const float* __restrict__ src, int ld) {
  int lane = threadIdx.x & 31, half = lane >> 4, m = lane & 15;
  const float* r = src + m * ld + half * 8;
  v16bf a;
#pragma unroll
  for (int j = 0; j < 8; ++j) { a[j] = (__bf16)r[j]; a[8 + j] = (__bf16)0.f; }
  return a;
}
__device__ __forceinline__ v16bf frag_a_bf16(const __bf16* src, int ld) {
  int lane = threadIdx.x & 31, half = lane >> 4, m = lane & 15;
  const __bf16* r = src + m * ld + half * 8;
  v16bf a;
#pragma unroll
  for (int j = 0; j < 8; ++j) { a[j] = r[j]; a[8 + j] = r[16 + j]; }
  return a;
}
// B-matrix (32x16, 16-bit), assumed to mirror A's K ordering with N across lanes.
__device__ __forceinline__ v16bf frag_b_f32(const float* __restrict__ src, int ld) {
  int lane = threadIdx.x & 31, half = lane >> 4, n = lane & 15;
  v16bf b;
#pragma unroll
  for (int j = 0; j < 8; ++j) {
    b[j]     = (__bf16)src[(half * 8 + j) * ld + n];
    b[8 + j] = (__bf16)src[(16 + half * 8 + j) * ld + n];
  }
  return b;
}
__device__ __forceinline__ v16bf frag_b_f32_k16(const float* __restrict__ src, int ld) {
  int lane = threadIdx.x & 31, half = lane >> 4, n = lane & 15;
  v16bf b;
#pragma unroll
  for (int j = 0; j < 8; ++j) {
    b[j]     = (__bf16)src[(half * 8 + j) * ld + n]; // rows 0..15 valid
    b[8 + j] = (__bf16)0.f;                          // rows 16..31 zero pad
  }
  return b;
}
__device__ __forceinline__ v16bf frag_b_bf16(const __bf16* src, int ld) {
  int lane = threadIdx.x & 31, half = lane >> 4, n = lane & 15;
  v16bf b;
#pragma unroll
  for (int j = 0; j < 8; ++j) {
    b[j]     = src[(half * 8 + j) * ld + n];
    b[8 + j] = src[(16 + half * 8 + j) * ld + n];
  }
  return b;
}
// D (16x16 f32): vgpr r = (M=r, N=lane) lanes 0-15, (M=r+8, N=lane-16) lanes 16-31
__device__ __forceinline__ void store_d(float* dst, int ld, v8f d) {
  int lane = threadIdx.x & 31, half = lane >> 4, n = lane & 15;
#pragma unroll
  for (int r = 0; r < 8; ++r) dst[(r + 8 * half) * ld + n] = d[r];
}

__device__ __forceinline__ float gelu_exact(float v) {
  return 0.5f * v * (1.f + erff(v * 0.70710678118654752f));
}

// ---------------------------------------------------------------------------
// Twiddle tables (bf16), built on device each launch. Layout in `tabs`:
//   FzT  [64z][16q]  @0     forward-z B matrix (q: Re k3 0..7 | Im k3 0..7)
//   FyA  [32r][64]   @1024  forward-y/x A matrix (rows: Re r 0..15 | Im r 0..15)
//   GxA  [128][32]   @3072  inverse-x/y A matrix (rows: Re x 0..63 | Im), K-pad 32
//   GzB  [32][64z]   @7168  inverse-z B matrix, K-pad rows 16..31 zero
// Mode index map: r<8 -> k=r ; r>=8 -> k=48+r (i.e. 56..63).
// ---------------------------------------------------------------------------
__global__ void k_init_tables(__bf16* tabs) {
  const float W0 = 6.28318530717958647692f / 64.f;
  int t = threadIdx.x;
  for (int i = t; i < 64 * 16; i += 256) { // FzT
    int z = i >> 4, q = i & 15;
    int ph = ((q & 7) * z) & 63;
    float th = W0 * (float)ph;
    tabs[i] = (__bf16)(((q < 8) ? cosf(th) : -sinf(th)) * 0.125f);
  }
  for (int i = t; i < 32 * 64; i += 256) { // FyA
    int r = i >> 6, y = i & 63;
    int rr = r & 15;
    int k = (rr < 8) ? rr : 48 + rr;
    float th = W0 * (float)((k * y) & 63);
    tabs[1024 + i] = (__bf16)(((r < 16) ? cosf(th) : -sinf(th)) * 0.125f);
  }
  for (int i = t; i < 128 * 32; i += 256) { // GxA (e^{+i th})
    int m = i >> 5, c = i & 31;
    float v = 0.f;
    if (c < 16) {
      int x = m & 63;
      int k = (c < 8) ? c : 48 + c;
      float th = W0 * (float)((k * x) & 63);
      v = ((m < 64) ? cosf(th) : sinf(th)) * 0.125f;
    }
    tabs[3072 + i] = (__bf16)v;
  }
  for (int i = t; i < 32 * 64; i += 256) { // GzB
    int q = i >> 6, z = i & 63;
    float v = 0.f;
    if (q < 16) {
      int k = q & 7;
      float cq = (k == 0) ? 0.125f : 0.25f; // DC once, others doubled (Hermitian)
      float th = W0 * (float)((k * z) & 63);
      v = (q < 8) ? cq * cosf(th) : -cq * sinf(th);
    }
    tabs[7168 + i] = (__bf16)v;
  }
}

// ---- lifting 1x1 conv: h[b,o,s] = pw[o,i] x[b,i,s] + pb[o] --------------------
__global__ void k_lift(const float* __restrict__ x, const float* __restrict__ pw,
                       const float* __restrict__ pb, float* __restrict__ h) {
  int idx = blockIdx.x * 256 + threadIdx.x;
  int b = idx >> 18, s = idx & (NVOX - 1);
  float in[9];
#pragma unroll
  for (int i = 0; i < 9; ++i) in[i] = x[(size_t)(b * 9 + i) * NVOX + s];
#pragma unroll
  for (int o = 0; o < 32; ++o) {
    float acc = pb[o];
#pragma unroll
    for (int i = 0; i < 9; ++i) acc += pw[o * 9 + i] * in[i];
    h[(size_t)(b * 32 + o) * NVOX + s] = acc;
  }
}

// ---- stage 1: forward DFT along z (64 -> 8 complex), real input ---------------
// per-wave slice (b,c,x): D[64y,16q] = H[64y,64z] x FzT[64z,16q]
__global__ void k_fwd_z(const float* __restrict__ h, float* __restrict__ T1,
                        const __bf16* __restrict__ tabs) {
  int wave = (blockIdx.x * blockDim.x + threadIdx.x) >> 5; // (b*32+c)*64 + x
  const __bf16* FzT = tabs;
  v16bf b0 = frag_b_bf16(FzT, 16);
  v16bf b1 = frag_b_bf16(FzT + 32 * 16, 16);
  const float* base = h + (size_t)wave * 4096;
  float* out = T1 + (size_t)wave * 1024;
#pragma unroll
  for (int mt = 0; mt < 4; ++mt) {
    v8f acc = vzero();
    acc = wmma_bf16(frag_a_f32(base + mt * 16 * 64, 64), b0, acc);
    acc = wmma_bf16(frag_a_f32(base + mt * 16 * 64 + 32, 64), b1, acc);
    store_d(out + mt * 16 * 16, 16, acc);
  }
}

// ---- stage 2: forward DFT along y (64 -> 16 complex modes), complex -----------
// per-wave slice (b,c,x): D[32,16] = FyA[32,64] x T1[64y,16q]; recombine re/im.
__global__ void k_fwd_y(const float* __restrict__ T1, float* __restrict__ T2,
                        const __bf16* __restrict__ tabs) {
  __shared__ float L[8][32][17];
  int wv = threadIdx.x >> 5;
  int wave = (blockIdx.x * blockDim.x + threadIdx.x) >> 5; // (b,c,x)
  const __bf16* FyA = tabs + 1024;
  const float* src = T1 + (size_t)wave * 1024;
  float* dst = T2 + (size_t)wave * 256;
#pragma unroll
  for (int mt = 0; mt < 2; ++mt) {
    v8f acc = vzero();
    acc = wmma_bf16(frag_a_bf16(FyA + mt * 16 * 64, 64), frag_b_f32(src, 16), acc);
    acc = wmma_bf16(frag_a_bf16(FyA + mt * 16 * 64 + 32, 64),
                    frag_b_f32(src + 32 * 16, 16), acc);
    store_d(&L[wv][mt * 16][0], 17, acc);
  }
  __syncthreads();
  int lane = threadIdx.x & 31;
#pragma unroll
  for (int t = 0; t < 8; ++t) {
    int o = t * 32 + lane; // r2*16 + q2
    int r = o >> 4, q = o & 15, k = q & 7;
    float v = (q < 8) ? (L[wv][r][k] - L[wv][16 + r][8 + k])
                      : (L[wv][r][8 + k] + L[wv][16 + r][k]);
    dst[o] = v;
  }
}

// ---- stage 3: forward DFT along x (64 -> 16 complex modes) --------------------
// per-wave slice (b,c,r2): rows are x-strided (stride 256 floats) in T2.
__global__ void k_fwd_x(const float* __restrict__ T2, float* __restrict__ T3,
                        const __bf16* __restrict__ tabs) {
  __shared__ float L[8][32][17];
  int wv = threadIdx.x >> 5;
  int wave = (blockIdx.x * blockDim.x + threadIdx.x) >> 5; // (b*32+c)*16 + r2
  int b = wave >> 9, c = (wave >> 4) & 31, r2 = wave & 15;
  const __bf16* FxA = tabs + 1024; // same matrix as FyA
  const float* src = T2 + (size_t)(b * 32 + c) * 16384 + r2 * 16;
  float* dst = T3 + (size_t)(b * 32 + c) * 4096 + r2 * 16;
#pragma unroll
  for (int mt = 0; mt < 2; ++mt) {
    v8f acc = vzero();
    acc = wmma_bf16(frag_a_bf16(FxA + mt * 16 * 64, 64), frag_b_f32(src, 256), acc);
    acc = wmma_bf16(frag_a_bf16(FxA + mt * 16 * 64 + 32, 64),
                    frag_b_f32(src + 32 * 256, 256), acc);
    store_d(&L[wv][mt * 16][0], 17, acc);
  }
  __syncthreads();
  int lane = threadIdx.x & 31;
#pragma unroll
  for (int t = 0; t < 8; ++t) {
    int o = t * 32 + lane; // r1*16 + q3
    int r = o >> 4, q = o & 15, k = q & 7;
    float v = (q < 8) ? (L[wv][r][k] - L[wv][16 + r][8 + k])
                      : (L[wv][r][8 + k] + L[wv][16 + r][k]);
    dst[r * 256 + q] = v;
  }
}

// ---- stage 4: spectral channel mix (batched tiny complex GEMM, VALU f32) ------
__global__ void k_specmul(const float* __restrict__ T3, float* __restrict__ T4,
                          const float* __restrict__ fw, int layer) {
  int idx = blockIdx.x * 256 + threadIdx.x; // (b*32+o)*2048 + mode
  int mode = idx & 2047, bo = idx >> 11, o = bo & 31, b = bo >> 5;
  int r1 = mode >> 7, r2 = (mode >> 3) & 15, k3 = mode & 7;
  int corner = (r1 >= 8 ? 1 : 0) + (r2 >= 8 ? 2 : 0);
  int m1 = r1 & 7, m2 = r2 & 7;
  const float* wb = fw + (size_t)layer * 4194304 + (size_t)corner * 1048576 +
                    o * 1024 + m1 * 128 + m2 * 16 + k3 * 2;
  const float* xb = T3 + (size_t)b * 131072 + r1 * 256 + r2 * 16 + k3;
  float ar = 0.f, ai = 0.f;
#pragma unroll 4
  for (int i = 0; i < 32; ++i) {
    float xr = xb[i * 4096], xi = xb[i * 4096 + 8];
    float wr = wb[(size_t)i * 32768], wi = wb[(size_t)i * 32768 + 1];
    ar += xr * wr - xi * wi;
    ai += xr * wi + xi * wr;
  }
  float* ob = T4 + (size_t)b * 131072 + o * 4096 + r1 * 256 + r2 * 16 + k3;
  ob[0] = ar;
  ob[8] = ai;
}

// ---- stage 5: inverse DFT along x (16 modes -> 64), complex -------------------
// per-wave slice (b,c,r2): D[128,16] = GxA[128,16(pad32)] x T4[16r1,16q]
__global__ void k_inv_x(const float* __restrict__ T4, float* __restrict__ T5,
                        const __bf16* __restrict__ tabs) {
  __shared__ float L[4][128][17];
  int wv = threadIdx.x >> 5;
  int wave = (blockIdx.x * blockDim.x + threadIdx.x) >> 5; // (b*32+c)*16 + r2
  int b = wave >> 9, c = (wave >> 4) & 31, r2 = wave & 15;
  const __bf16* GxA = tabs + 3072;
  const float* src = T4 + (size_t)(b * 32 + c) * 4096 + r2 * 16;
  v16bf bb = frag_b_f32_k16(src, 256);
#pragma unroll
  for (int mt = 0; mt < 8; ++mt) {
    v8f acc = vzero();
    acc = wmma_bf16(frag_a_bf16(GxA + mt * 16 * 32, 32), bb, acc);
    store_d(&L[wv][mt * 16][0], 17, acc);
  }
  __syncthreads();
  float* dst = T5 + (size_t)(b * 32 + c) * 16384 + r2 * 16;
  int lane = threadIdx.x & 31;
#pragma unroll
  for (int t = 0; t < 32; ++t) {
    int o = t * 32 + lane; // x*16 + q
    int xx = o >> 4, q = o & 15, k = q & 7;
    float v = (q < 8) ? (L[wv][xx][k] - L[wv][64 + xx][8 + k])
                      : (L[wv][xx][8 + k] + L[wv][64 + xx][k]);
    dst[(size_t)xx * 256 + q] = v;
  }
}

// ---- stage 6: inverse DFT along y (16 modes -> 64), complex -------------------
// per-wave slice (b,c,x): contiguous 16x16 input tile in T5.
__global__ void k_inv_y(const float* __restrict__ T5, float* __restrict__ T6,
                        const __bf16* __restrict__ tabs) {
  __shared__ float L[4][128][17];
  int wv = threadIdx.x >> 5;
  int wave = (blockIdx.x * blockDim.x + threadIdx.x) >> 5; // (b*32+c)*64 + x
  const __bf16* GyA = tabs + 3072; // same matrix as GxA
  const float* src = T5 + (size_t)wave * 256;
  v16bf bb = frag_b_f32_k16(src, 16);
#pragma unroll
  for (int mt = 0; mt < 8; ++mt) {
    v8f acc = vzero();
    acc = wmma_bf16(frag_a_bf16(GyA + mt * 16 * 32, 32), bb, acc);
    store_d(&L[wv][mt * 16][0], 17, acc);
  }
  __syncthreads();
  float* dst = T6 + (size_t)wave * 1024;
  int lane = threadIdx.x & 31;
#pragma unroll
  for (int t = 0; t < 32; ++t) {
    int o = t * 32 + lane; // y*16 + q
    int yy = o >> 4, q = o & 15, k = q & 7;
    float v = (q < 8) ? (L[wv][yy][k] - L[wv][64 + yy][8 + k])
                      : (L[wv][yy][8 + k] + L[wv][64 + yy][k]);
    dst[o] = v;
  }
}

// ---- stage 7: inverse real DFT along z (8 complex -> 64 real) -----------------
// per-wave slice (b,c,x): XF[64y,64z] = T6[64y,16q(pad32)] x GzB[32,64z]
__global__ void k_inv_z(const float* __restrict__ T6, float* __restrict__ XF,
                        const __bf16* __restrict__ tabs) {
  int wave = (blockIdx.x * blockDim.x + threadIdx.x) >> 5; // (b*32+c)*64 + x
  const __bf16* GzB = tabs + 7168;
  v16bf b0 = frag_b_bf16(GzB + 0, 64);
  v16bf b1 = frag_b_bf16(GzB + 16, 64);
  v16bf b2 = frag_b_bf16(GzB + 32, 64);
  v16bf b3 = frag_b_bf16(GzB + 48, 64);
  const float* src = T6 + (size_t)wave * 1024;
  float* dst = XF + (size_t)wave * 4096;
#pragma unroll
  for (int mt = 0; mt < 4; ++mt) {
    v16bf aa = frag_a_f32_k16(src + mt * 16 * 16, 16);
    v8f d0 = vzero(), d1 = vzero(), d2 = vzero(), d3 = vzero();
    d0 = wmma_bf16(aa, b0, d0);
    d1 = wmma_bf16(aa, b1, d1);
    d2 = wmma_bf16(aa, b2, d2);
    d3 = wmma_bf16(aa, b3, d3);
    store_d(dst + mt * 1024 + 0, 64, d0);
    store_d(dst + mt * 1024 + 16, 64, d1);
    store_d(dst + mt * 1024 + 32, 64, d2);
    store_d(dst + mt * 1024 + 48, 64, d3);
  }
}

// ---- stage 8: fused pointwise conv + residual + exact GELU --------------------
// block = (b, 128-point spatial tile): Hbuf := gelu(XF + cw[l] @ Hbuf + cb[l])
__global__ void k_conv_gelu(float* __restrict__ Hbuf, const float* __restrict__ XF,
                            const float* __restrict__ cw,
                            const float* __restrict__ cb, int layer) {
  __shared__ __bf16 Lh[32][128];
  int blk = blockIdx.x;
  int b = blk >> 11;
  int s0 = (blk & 2047) * 128;
  int tid = threadIdx.x;
  for (int i = tid; i < 4096; i += 256) { // stage 32ch x 128pts, coalesced
    int ch = i >> 7, off = i & 127;
    Lh[ch][off] = (__bf16)Hbuf[(size_t)(b * 32 + ch) * NVOX + s0 + off];
  }
  __syncthreads();
  const float* A = cw + layer * 1024; // [32o][32i] row-major
  int wv = tid >> 5, lane = tid & 31;
  v16bf bb = frag_b_bf16(&Lh[0][wv * 16], 128);
  v8f d0 = vzero(), d1 = vzero();
  d0 = wmma_bf16(frag_a_f32(A, 32), bb, d0);
  d1 = wmma_bf16(frag_a_f32(A + 16 * 32, 32), bb, d1);
  int half = lane >> 4, n = lane & 15;
  int s = s0 + wv * 16 + n;
#pragma unroll
  for (int r = 0; r < 8; ++r) {
    int o = r + 8 * half;
    float v = d0[r] + cb[layer * 32 + o] + XF[(size_t)(b * 32 + o) * NVOX + s];
    Hbuf[(size_t)(b * 32 + o) * NVOX + s] = gelu_exact(v);
    int o2 = o + 16;
    float v2 = d1[r] + cb[layer * 32 + o2] + XF[(size_t)(b * 32 + o2) * NVOX + s];
    Hbuf[(size_t)(b * 32 + o2) * NVOX + s] = gelu_exact(v2);
  }
}

// ---- projection 1x1 conv (32 -> 1) -------------------------------------------
__global__ void k_proj(const float* __restrict__ h, const float* __restrict__ qw,
                       const float* __restrict__ qb, float* __restrict__ out) {
  int idx = blockIdx.x * 256 + threadIdx.x;
  int b = idx >> 18, s = idx & (NVOX - 1);
  float acc = qb[0];
#pragma unroll
  for (int i = 0; i < 32; ++i) acc += qw[i] * h[(size_t)(b * 32 + i) * NVOX + s];
  out[idx] = acc;
}

// ---------------------------------------------------------------------------
extern "C" void kernel_launch(void* const* d_in, const int* in_sizes, int n_in,
                              void* d_out, int out_size, void* d_ws, size_t ws_size,
                              hipStream_t stream) {
  (void)in_sizes; (void)n_in; (void)out_size; (void)ws_size;
  const float* x  = (const float*)d_in[0];
  const float* pw = (const float*)d_in[1];
  const float* pb = (const float*)d_in[2];
  const float* fw = (const float*)d_in[3];
  const float* cw = (const float*)d_in[4];
  const float* cb = (const float*)d_in[5];
  const float* qw = (const float*)d_in[6];
  const float* qb = (const float*)d_in[7];
  float* out = (float*)d_out;

  // workspace carve-up (floats)
  float* W    = (float*)d_ws;
  float* Hbuf = W;                  // [4][32][64^3]          33,554,432 f
  float* XF   = Hbuf + 33554432;    // [4][32][64^3]          33,554,432 f
  float* BA   = XF + 33554432;      // T1 / T6                 8,388,608 f
  float* BB   = BA + 8388608;       // T2 / T5                 2,097,152 f
  float* T3   = BB + 2097152;       //                           524,288 f
  float* T4   = T3 + 524288;        //                           524,288 f
  __bf16* tabs = (__bf16*)(T4 + 524288); // 9216 bf16 twiddles

  k_init_tables<<<1, 256, 0, stream>>>(tabs);
  k_lift<<<4096, 256, 0, stream>>>(x, pw, pb, Hbuf);
  for (int l = 0; l < 4; ++l) {
    k_fwd_z<<<1024, 256, 0, stream>>>(Hbuf, BA, tabs);
    k_fwd_y<<<1024, 256, 0, stream>>>(BA, BB, tabs);
    k_fwd_x<<<256, 256, 0, stream>>>(BB, T3, tabs);
    k_specmul<<<1024, 256, 0, stream>>>(T3, T4, fw, l);
    k_inv_x<<<512, 128, 0, stream>>>(T4, BB, tabs);
    k_inv_y<<<2048, 128, 0, stream>>>(BB, BA, tabs);
    k_inv_z<<<1024, 256, 0, stream>>>(BA, XF, tabs);
    k_conv_gelu<<<8192, 256, 0, stream>>>(Hbuf, XF, cw, cb, l);
  }
  k_proj<<<4096, 256, 0, stream>>>(Hbuf, qw, qb, out);
}
